// EchoStateNetwork_41042707480746
// MI455X (gfx1250) — matrix operand order repaired and checked by reading, not compiled, not run on hardware
//
#include <hip/hip_runtime.h>
#include <math.h>

#define RDIM 2048
#define IDIM 128
#define BDIM 64
#define TDIM 2048
#define ODIM 128
#define NWG  64
#define WAVES_PER_WG 8

typedef __attribute__((ext_vector_type(8)))  __bf16 v8bf;
typedef __attribute__((ext_vector_type(16))) __bf16 v16bf;
typedef __attribute__((ext_vector_type(8)))  float  v8f;
typedef __attribute__((ext_vector_type(4)))  float  v4f;

// f32 -> bf16 via native convert (RNE on CDNA5: v_cvt_pk_bf16_f32 class)
static __device__ __forceinline__ __bf16 f2bf(float f) {
  return (__bf16)f;
}

// WGP-scope prefetch (scope field 0 = pull into all cache levels incl. WGP$).
// __builtin_prefetch only reaches SE/SYS scope, which skips the WGP cache.
static __device__ __forceinline__ void prefetch_wgp(const void* p) {
  asm volatile("global_prefetch_b8 %0, off" :: "v"(p));
}

// ---------------------------------------------------------------------------
// Prep: wres/win f32 -> bf16, zero h0, zero sync counter (re-done every call)
// ---------------------------------------------------------------------------
__global__ void esn_prep(const float* __restrict__ wres, const float* __restrict__ win,
                         __bf16* __restrict__ wres_b, __bf16* __restrict__ win_b,
                         __bf16* __restrict__ h0, unsigned* __restrict__ cnt) {
  size_t i = (size_t)blockIdx.x * blockDim.x + threadIdx.x;
  if (i < (size_t)RDIM * RDIM) wres_b[i] = f2bf(wres[i]);
  if (i < (size_t)RDIM * IDIM) win_b[i]  = f2bf(win[i]);
  if (i < (size_t)BDIM * RDIM) h0[i]     = __builtin_bit_cast(__bf16, (unsigned short)0);
  if (i == 0) *cnt = 0u;
}

// ---------------------------------------------------------------------------
// Persistent recurrence kernel: all T steps, device-wide barrier per step.
// 64 WGs x 256 threads = 512 waves; one wave per 16x16 tile of h (4 x 128).
// ---------------------------------------------------------------------------
__global__ void __launch_bounds__(256) esn_steps(
    const float* __restrict__ x,
    const __bf16* __restrict__ wres_b, const __bf16* __restrict__ win_b,
    __bf16* __restrict__ h0, __bf16* __restrict__ h1,
    float* __restrict__ hfinal, unsigned* __restrict__ cnt)
{
  const int tid  = threadIdx.x;
  const int lane = tid & 31;
  const int lm   = lane & 15;
  const int hi   = lane >> 4;                 // 0: lanes 0-15, 1: lanes 16-31
  const int wave = blockIdx.x * WAVES_PER_WG + (tid >> 5);
  const int m0 = (wave & 3) * 16;             // batch-tile (4 tiles of 16 over B=64)
  const int n0 = (wave >> 2) * 16;            // reservoir-tile (128 tiles over R=2048)

  const int koffA = hi ? 8 : 0;               // 16-bit A layout: chunks K+{0..7},K+{16..23} / +8
  const int koffB = hi ? 8 : 0;               // 16-bit B layout: chunks K+{0..7},K+{16..23} / +8

  const __bf16* browR = wres_b + (size_t)(n0 + lm) * RDIM;  // B col n = wres row n (contig K!)
  const __bf16* browI = win_b  + (size_t)(n0 + lm) * IDIM;
  const float*  xbase = x + (size_t)(m0 + lm) * TDIM * IDIM;
  const int col = n0 + lm;

  for (int t = 0; t < TDIM; ++t) {
    const __bf16* hp = (t & 1) ? h1 : h0;
    __bf16*       hn = (t & 1) ? h0 : h1;

    v8f acc0 = {};
    v8f acc1 = {};

    // ---- reservoir term: h_{t-1} @ wres^T, K = 2048, 2 independent chains ----
    const __bf16* arow = hp + (size_t)(m0 + lm) * RDIM;
#pragma unroll 2
    for (int k = 0; k < RDIM; k += 64) {
      // WGP-scope prefetch ~2KB ahead (stays inside the workspace allocation)
      prefetch_wgp(browR + k + 1024);
      prefetch_wgp(arow  + k + 1024);
      {
        v8bf a0 = *(const v8bf*)(arow + k + koffA);
        v8bf a1 = *(const v8bf*)(arow + k + koffA + 16);
        v8bf b0 = *(const v8bf*)(browR + k + koffB);
        v8bf b1 = *(const v8bf*)(browR + k + koffB + 16);
        v16bf A  = __builtin_shufflevector(a0, a1, 0,1,2,3,4,5,6,7,8,9,10,11,12,13,14,15);
        v16bf Bv = __builtin_shufflevector(b0, b1, 0,1,2,3,4,5,6,7,8,9,10,11,12,13,14,15);
        acc0 = __builtin_amdgcn_wmma_f32_16x16x32_bf16(false, A, false, Bv,
                                                       (short)0, acc0, false, false);
      }
      {
        const int k2 = k + 32;
        v8bf a0 = *(const v8bf*)(arow + k2 + koffA);
        v8bf a1 = *(const v8bf*)(arow + k2 + koffA + 16);
        v8bf b0 = *(const v8bf*)(browR + k2 + koffB);
        v8bf b1 = *(const v8bf*)(browR + k2 + koffB + 16);
        v16bf A  = __builtin_shufflevector(a0, a1, 0,1,2,3,4,5,6,7,8,9,10,11,12,13,14,15);
        v16bf Bv = __builtin_shufflevector(b0, b1, 0,1,2,3,4,5,6,7,8,9,10,11,12,13,14,15);
        acc1 = __builtin_amdgcn_wmma_f32_16x16x32_bf16(false, A, false, Bv,
                                                       (short)0, acc1, false, false);
      }
    }

    // ---- input term: x_t @ win^T, K = 128 (native f32 -> bf16 converts) ----
    const float* xrow = xbase + (size_t)t * IDIM;
#pragma unroll
    for (int k = 0; k < IDIM; k += 32) {
      const int k2 = k + koffA;
      v4f f0 = *(const v4f*)(xrow + k2);
      v4f f1 = *(const v4f*)(xrow + k2 + 4);
      v4f f2 = *(const v4f*)(xrow + k2 + 16);
      v4f f3 = *(const v4f*)(xrow + k2 + 20);
      v16bf A;
#pragma unroll
      for (int i = 0; i < 4; ++i) {
        A[i]      = f2bf(f0[i]);
        A[i + 4]  = f2bf(f1[i]);
        A[i + 8]  = f2bf(f2[i]);
        A[i + 12] = f2bf(f3[i]);
      }
      v8bf b0 = *(const v8bf*)(browI + k + koffB);
      v8bf b1 = *(const v8bf*)(browI + k + koffB + 16);
      v16bf Bv = __builtin_shufflevector(b0, b1, 0,1,2,3,4,5,6,7,8,9,10,11,12,13,14,15);
      if (((k >> 5) & 1) == 0)
        acc0 = __builtin_amdgcn_wmma_f32_16x16x32_bf16(false, A, false, Bv,
                                                       (short)0, acc0, false, false);
      else
        acc1 = __builtin_amdgcn_wmma_f32_16x16x32_bf16(false, A, false, Bv,
                                                       (short)0, acc1, false, false);
    }

    // ---- tanh + write h_t (C/D layout: vgpr v -> M = v + 8*hi, N = lane&15) ----
    __bf16* hncol = hn + col;
#pragma unroll
    for (int v = 0; v < 8; ++v) {
      float z;
#if __has_builtin(__builtin_amdgcn_tanhf)
      z = __builtin_amdgcn_tanhf(acc0[v] + acc1[v]);
#else
      z = tanhf(acc0[v] + acc1[v]);
#endif
      const int row = m0 + v + (hi ? 8 : 0);
      hncol[(size_t)row * RDIM] = f2bf(z);
      if (t == TDIM - 1) hfinal[(size_t)row * RDIM + col] = z;
    }

    // ---- device-wide step barrier (release arrive, acquire spin) ----
    __threadfence();
    __syncthreads();
    if (tid == 0) {
      __hip_atomic_fetch_add(cnt, 1u, __ATOMIC_RELEASE, __HIP_MEMORY_SCOPE_AGENT);
      const unsigned target = (unsigned)(t + 1) * (unsigned)NWG;
      while (__hip_atomic_load(cnt, __ATOMIC_ACQUIRE, __HIP_MEMORY_SCOPE_AGENT) < target)
        __builtin_amdgcn_s_sleep(2);
    }
    __syncthreads();
    __threadfence();
  }
}

// ---------------------------------------------------------------------------
// Readout: out[m,o] = b[o] + x_last[m,:]*w_out[o,0:I] + h_final[m,:]*w_out[o,I:]
// ---------------------------------------------------------------------------
__global__ void esn_readout(const float* __restrict__ x, const float* __restrict__ hfinal,
                            const float* __restrict__ wout, const float* __restrict__ bout,
                            float* __restrict__ out) {
  int idx = blockIdx.x * blockDim.x + threadIdx.x;
  if (idx >= BDIM * ODIM) return;
  int m = idx / ODIM, o = idx % ODIM;
  const float* xl = x + ((size_t)m * TDIM + (TDIM - 1)) * IDIM;
  const float* wr = wout + (size_t)o * (RDIM + IDIM);
  float s = bout[o];
  for (int i = 0; i < IDIM; ++i) s = fmaf(xl[i], wr[i], s);
  const float* hm = hfinal + (size_t)m * RDIM;
  for (int r = 0; r < RDIM; ++r) s = fmaf(hm[r], wr[IDIM + r], s);
  out[(size_t)m * ODIM + o] = s;
}

// ---------------------------------------------------------------------------
extern "C" void kernel_launch(void* const* d_in, const int* in_sizes, int n_in,
                              void* d_out, int out_size, void* d_ws, size_t ws_size,
                              hipStream_t stream) {
  const float* x    = (const float*)d_in[0];   // [B,T,I]
  const float* win  = (const float*)d_in[1];   // [R,I]
  const float* wres = (const float*)d_in[2];   // [R,R]
  const float* wout = (const float*)d_in[3];   // [O,R+I]
  const float* bout = (const float*)d_in[4];   // [O]
  float* out = (float*)d_out;

  char* ws = (char*)d_ws;
  const size_t szWres = (size_t)RDIM * RDIM * 2;   // 8 MB bf16
  const size_t szWin  = (size_t)RDIM * IDIM * 2;   // 512 KB
  const size_t szH    = (size_t)BDIM * RDIM * 2;   // 256 KB
  const size_t szHF   = (size_t)BDIM * RDIM * 4;   // 512 KB
  __bf16*   wres_b = (__bf16*)ws;
  __bf16*   win_b  = (__bf16*)(ws + szWres);
  __bf16*   h0     = (__bf16*)(ws + szWres + szWin);
  __bf16*   h1     = (__bf16*)(ws + szWres + szWin + szH);
  float*    hfin   = (float*) (ws + szWres + szWin + 2 * szH);
  unsigned* cnt    = (unsigned*)(ws + szWres + szWin + 2 * szH + szHF);

  const int prepBlocks = (RDIM * RDIM + 255) / 256;
  esn_prep<<<prepBlocks, 256, 0, stream>>>(wres, win, wres_b, win_b, h0, cnt);
  esn_steps<<<NWG, 256, 0, stream>>>(x, wres_b, win_b, h0, h1, hfin, cnt);
  esn_readout<<<(BDIM * ODIM + 255) / 256, 256, 0, stream>>>(x, hfin, wout, bout, out);
}
